// SparseAutoencoder_12378095747818
// MI455X (gfx1250) — compile-verified, matrix operand order checked by reading, not run on hardware
//
#include <hip/hip_runtime.h>
#include <stdint.h>

// ---------- CDNA5 vector types ----------
typedef __attribute__((ext_vector_type(16))) __bf16 v16bf;
typedef __attribute__((ext_vector_type(8)))  float  v8f;
typedef __attribute__((ext_vector_type(4)))  unsigned u32x4;
typedef __attribute__((ext_vector_type(4)))  float    f32x4;

#define D_VIT 1024
#define D_SAE 16384
#define BATCH 8192
#define TOPK  32

// GEMM tiling: workgroup = 8 waves (wave32), macro tile 128(M) x 128(N), K-step 32
// waves arranged 4(M) x 2(N); wave tile 32(M) x 64(N) = 2x4 WMMA tiles
#define TM 128
#define TN 128
#define TK 32

// ---------- bf16 split helpers ----------
__device__ __forceinline__ unsigned short f32_to_bf16_rne(float f) {
  unsigned u = __builtin_bit_cast(unsigned, f);
  unsigned r = u + 0x7FFFu + ((u >> 16) & 1u);
  return (unsigned short)(r >> 16);
}
__device__ __forceinline__ float bf16_to_f32(unsigned short h) {
  unsigned u = ((unsigned)h) << 16;
  return __builtin_bit_cast(float, u);
}
__device__ __forceinline__ void split_bf16(float f, unsigned short& hi, unsigned short& lo) {
  hi = f32_to_bf16_rne(f);
  lo = f32_to_bf16_rne(f - bf16_to_f32(hi));
}
__device__ __forceinline__ unsigned pack2(unsigned short a, unsigned short b) {
  return (unsigned)a | ((unsigned)b << 16);
}

union FragCast { u32x4 q[2]; v16bf v; };

// Load a 16x32 bf16 A/B fragment from LDS (row-contiguous 64B rows) into the
// ISA-specified per-lane VGPR layout:
//   lanes 0-15 : K0..7  = row bytes [0..15],  K16..23 = bytes [32..47]
//   lanes 16-31: K8..15 = row bytes [16..31], K24..31 = bytes [48..63]
__device__ __forceinline__ v16bf load_frag(const unsigned* base, int rowIdx, int halfsel) {
  FragCast fc;
  const u32x4* p = (const u32x4*)(base + rowIdx * 16 + halfsel * 4);
  fc.q[0] = p[0];   // first 16B chunk
  fc.q[1] = p[2];   // +32 bytes
  return fc.v;
}

// ============================================================================
// Kernel 1: h_pre = x @ W_enc + b_enc   (bf16x3 split on v_wmma_f32_16x16x32_bf16)
// writes h_pre into the f_x region of d_out (sparsified in place by kernel 2)
// ============================================================================
__launch_bounds__(256)
__global__ void sae_encoder_gemm(const float* __restrict__ x,
                                 const float* __restrict__ Wenc,
                                 const float* __restrict__ benc,
                                 float* __restrict__ hout) {
  __shared__ __align__(16) unsigned Ahi[TM * (TK / 2)];   // 8 KB  ([m][kpair])
  __shared__ __align__(16) unsigned Alo[TM * (TK / 2)];   // 8 KB
  __shared__ __align__(16) unsigned Bhi[TN * (TK / 2)];   // 8 KB  (B^T: [n][kpair])
  __shared__ __align__(16) unsigned Blo[TN * (TK / 2)];   // 8 KB

  const int tid     = threadIdx.x;
  const int lane    = tid & 31;
  const int wave    = tid >> 5;
  const int wm      = wave & 3;       // wave M block: wm*32
  const int wn      = wave >> 2;      // wave N block: wn*64
  const int lanelo  = lane & 15;
  const int halfsel = lane >> 4;      // 0 or 1

  const int n0 = blockIdx.x * TN;
  const int m0 = blockIdx.y * TM;

  v8f acc[2][4];
  #pragma unroll
  for (int mt = 0; mt < 2; ++mt)
    #pragma unroll
    for (int nt = 0; nt < 4; ++nt) acc[mt][nt] = {};

  // A-stage mapping: 2 threads per row, 16 consecutive f32 each
  const int ar = tid >> 1;     // 0..127
  const int ah = tid & 1;      // which 16-float half of the 32-float K-chunk

  for (int k0 = 0; k0 < D_VIT; k0 += TK) {
    __syncthreads();   // previous compute phase done

    // ---- global -> LDS: A tile [128 x 32] f32, split hi/lo bf16, packed ----
    const f32x4* agp = (const f32x4*)(x + (size_t)(m0 + ar) * D_VIT + k0 + ah * 16);
    f32x4 av[4];
    av[0] = agp[0]; av[1] = agp[1]; av[2] = agp[2]; av[3] = agp[3];

    if (k0 + TK < D_VIT) {  // speculative prefetch of next K-slab (global_prefetch_b8)
      __builtin_prefetch(x + (size_t)(m0 + ar) * D_VIT + (k0 + TK) + ah * 16, 0, 1);
      __builtin_prefetch(Wenc + (size_t)(k0 + TK + (tid >> 3)) * D_SAE + n0, 0, 1);
    }

    {
      unsigned hw[8], lw[8];
      #pragma unroll
      for (int q = 0; q < 4; ++q) {
        unsigned short h0, l0, h1, l1, h2, l2, h3, l3;
        split_bf16(av[q][0], h0, l0);
        split_bf16(av[q][1], h1, l1);
        split_bf16(av[q][2], h2, l2);
        split_bf16(av[q][3], h3, l3);
        hw[q * 2 + 0] = pack2(h0, h1); hw[q * 2 + 1] = pack2(h2, h3);
        lw[q * 2 + 0] = pack2(l0, l1); lw[q * 2 + 1] = pack2(l2, l3);
      }
      const int ab = ar * 16 + ah * 8;   // uint index; byte 64*ar+32*ah (16B aligned)
      *(u32x4*)(&Ahi[ab])     = u32x4{hw[0], hw[1], hw[2], hw[3]};
      *(u32x4*)(&Ahi[ab + 4]) = u32x4{hw[4], hw[5], hw[6], hw[7]};
      *(u32x4*)(&Alo[ab])     = u32x4{lw[0], lw[1], lw[2], lw[3]};
      *(u32x4*)(&Alo[ab + 4]) = u32x4{lw[4], lw[5], lw[6], lw[7]};
    }

    // ---- global -> LDS: B tile [32 x 128] f32, coalesced; two adjacent K-rows
    // per thread so the transposed store is a packed (k,k+1) dword ----
    #pragma unroll
    for (int i = 0; i < 2; ++i) {
      const int p  = tid + 256 * i;    // 0..511 K-row-pair x float4 work items
      const int kp = p >> 5;           // 0..15 -> k = 2*kp, 2*kp+1
      const int nn = (p & 31) * 4;     // 0..124
      const float* gb = Wenc + (size_t)(k0 + kp * 2) * D_SAE + n0 + nn;
      const f32x4 w0 = *(const f32x4*)gb;
      const f32x4 w1 = *(const f32x4*)(gb + D_SAE);
      #pragma unroll
      for (int j = 0; j < 4; ++j) {
        unsigned short h0, l0, h1, l1;
        split_bf16(w0[j], h0, l0);
        split_bf16(w1[j], h1, l1);
        Bhi[(nn + j) * 16 + kp] = pack2(h0, h1);   // K even in [15:0]
        Blo[(nn + j) * 16 + kp] = pack2(l0, l1);
      }
    }

    __syncthreads();   // tiles visible

    // ---- fragment loads ----
    v16bf af_hi[2], af_lo[2], bf_hi[4], bf_lo[4];
    #pragma unroll
    for (int mt = 0; mt < 2; ++mt) {
      const int r = wm * 32 + mt * 16 + lanelo;
      af_hi[mt] = load_frag(Ahi, r, halfsel);
      af_lo[mt] = load_frag(Alo, r, halfsel);
    }
    #pragma unroll
    for (int nt = 0; nt < 4; ++nt) {
      const int r = wn * 64 + nt * 16 + lanelo;
      bf_hi[nt] = load_frag(Bhi, r, halfsel);
      bf_lo[nt] = load_frag(Blo, r, halfsel);
    }

    // ---- bf16x3 split WMMA: hi*hi + hi*lo + lo*hi ----
    #pragma unroll
    for (int mt = 0; mt < 2; ++mt) {
      #pragma unroll
      for (int nt = 0; nt < 4; ++nt) {
        acc[mt][nt] = __builtin_amdgcn_wmma_f32_16x16x32_bf16(
            false, af_hi[mt], false, bf_hi[nt], (short)0, acc[mt][nt], false, false);
        acc[mt][nt] = __builtin_amdgcn_wmma_f32_16x16x32_bf16(
            false, af_hi[mt], false, bf_lo[nt], (short)0, acc[mt][nt], false, false);
        acc[mt][nt] = __builtin_amdgcn_wmma_f32_16x16x32_bf16(
            false, af_lo[mt], false, bf_hi[nt], (short)0, acc[mt][nt], false, false);
      }
    }
  }

  // ---- epilogue: + b_enc, store h_pre ----
  // C/D layout: VGPR r -> M = r (lanes 0-15) or 8+r (lanes 16-31); N = lane%16
  #pragma unroll
  for (int nt = 0; nt < 4; ++nt) {
    const int col = n0 + wn * 64 + nt * 16 + lanelo;
    const float be = benc[col];
    #pragma unroll
    for (int mt = 0; mt < 2; ++mt) {
      const int mbase = m0 + wm * 32 + mt * 16 + ((lane < 16) ? 0 : 8);
      #pragma unroll
      for (int r = 0; r < 8; ++r) {
        hout[(size_t)(mbase + r) * D_SAE + col] = acc[mt][nt][r] + be;
      }
    }
  }
}

// ============================================================================
// Kernel 2: per-row top-32, sparsify in place, emit (idx,val) lists to ws
// ============================================================================
__launch_bounds__(256)
__global__ void sae_topk(float* __restrict__ f,          // [BATCH][D_SAE] h_pre -> f_x
                         int* __restrict__ ws_idx,
                         float* __restrict__ ws_val) {
  __shared__ unsigned mask[D_SAE / 32];   // 2 KB selection bitmask
  __shared__ float tmaxv[256];
  __shared__ int   tmaxi[256];
  __shared__ int   s_winner;

  const int tid = threadIdx.x;
  const int row = blockIdx.x;
  float* rowp = f + (size_t)row * D_SAE;

  mask[tid] = 0u;
  mask[tid + 256] = 0u;

  // per-thread max over strided 64-element chunk (i % 256 == tid)
  float mv = -__builtin_inff(); int mi = -1;
  #pragma unroll 8
  for (int j = 0; j < 64; ++j) {
    const int i = tid + j * 256;
    const float v = rowp[i];
    if (v > mv) { mv = v; mi = i; }   // strict > keeps lowest index on ties
  }
  tmaxv[tid] = mv; tmaxi[tid] = mi;
  __syncthreads();

  for (int it = 0; it < TOPK; ++it) {
    if (tid < 32) {   // wave 0: argmax over the 256 candidates
      float bv = -__builtin_inff(); int bi = 0x7fffffff; int bt = 0;
      #pragma unroll
      for (int j2 = 0; j2 < 8; ++j2) {
        const int t2 = tid * 8 + j2;
        const float v = tmaxv[t2]; const int i2 = tmaxi[t2];
        if (v > bv || (v == bv && i2 < bi)) { bv = v; bi = i2; bt = t2; }
      }
      #pragma unroll
      for (int off = 16; off > 0; off >>= 1) {
        const float ov = __shfl_xor(bv, off, 32);
        const int   oi = __shfl_xor(bi, off, 32);
        const int   ot = __shfl_xor(bt, off, 32);
        if (ov > bv || (ov == bv && oi < bi)) { bv = ov; bi = oi; bt = ot; }
      }
      if (tid == 0) {
        s_winner = bt;
        mask[bi >> 5] |= (1u << (bi & 31));
        ws_idx[row * TOPK + it] = bi;
        ws_val[row * TOPK + it] = bv;
      }
    }
    __syncthreads();
    if (tid == s_winner) {   // only the winning thread rescans its chunk
      float nv = -__builtin_inff(); int ni = -1;
      for (int j = 0; j < 64; ++j) {
        const int i = tid + j * 256;
        if (mask[i >> 5] & (1u << (i & 31))) continue;
        const float v = rowp[i];
        if (v > nv) { nv = v; ni = i; }
      }
      tmaxv[tid] = nv; tmaxi[tid] = ni;
    }
    __syncthreads();
  }

  // sparsify in place: keep selected, zero the rest
  #pragma unroll 8
  for (int j = 0; j < 64; ++j) {
    const int i = tid + j * 256;
    const float v = rowp[i];
    const bool sel = (mask[i >> 5] >> (i & 31)) & 1u;
    rowp[i] = sel ? v : 0.0f;
  }
}

// ============================================================================
// Kernel 3: x_hat = f_x @ W_dec + b_dec as a 32-row gather (L2-resident W_dec)
// ============================================================================
__launch_bounds__(256)
__global__ void sae_decoder(const float* __restrict__ Wdec,
                            const float* __restrict__ bdec,
                            const int* __restrict__ ws_idx,
                            const float* __restrict__ ws_val,
                            float* __restrict__ xhat) {
  const int row = blockIdx.x;
  const int tid = threadIdx.x;          // 256 threads x float4 = 1024 cols
  f32x4 acc = *(const f32x4*)(bdec + tid * 4);
  #pragma unroll 4
  for (int i = 0; i < TOPK; ++i) {
    const int   idx = ws_idx[row * TOPK + i];
    const float v   = ws_val[row * TOPK + i];
    const f32x4 w = *(const f32x4*)(Wdec + (size_t)idx * D_VIT + tid * 4);
    acc += v * w;
  }
  *(f32x4*)(xhat + (size_t)row * D_VIT + tid * 4) = acc;
}

// ============================================================================
extern "C" void kernel_launch(void* const* d_in, const int* in_sizes, int n_in,
                              void* d_out, int out_size, void* d_ws, size_t ws_size,
                              hipStream_t stream) {
  const float* x    = (const float*)d_in[0];
  const float* Wenc = (const float*)d_in[1];
  const float* benc = (const float*)d_in[2];
  const float* Wdec = (const float*)d_in[3];
  const float* bdec = (const float*)d_in[4];
  // d_in[5] = k (==32), hardcoded as TOPK

  float* xhat = (float*)d_out;
  float* fx   = (float*)d_out + (size_t)BATCH * D_VIT;   // f_x region (h_pre scratch)

  int*   wsi = (int*)d_ws;                                                  // 1 MB
  float* wsv = (float*)((char*)d_ws + (size_t)BATCH * TOPK * sizeof(int));  // 1 MB

  dim3 g1(D_SAE / TN, BATCH / TM);   // 128 x 64 workgroups
  sae_encoder_gemm<<<g1, dim3(256), 0, stream>>>(x, Wenc, benc, fx);
  sae_topk<<<dim3(BATCH), dim3(256), 0, stream>>>(fx, wsi, wsv);
  sae_decoder<<<dim3(BATCH), dim3(256), 0, stream>>>(Wdec, bdec, wsi, wsv, xhat);
}
